// cha_third_conv_20040317403396
// MI455X (gfx1250) — compile-verified, matrix operand order checked by reading, not run on hardware
//
#include <hip/hip_runtime.h>
#include <hip/hip_bf16.h>

// ---------------------------------------------------------------------------
// input  : (4, 256, 56, 56) fp32      weight : (256, 5248) fp32
// bias   : (256,) fp32                pm2 : (36,2) i32   pcm : (120,2) i32
// out    : (4, 256, 56, 56) fp32
// GEMM: M=256, K=5248 (32 groups x 164, padded to 32x192), N=12544
// Compute-bound (AI ~1100 FLOP/B) -> bf16 WMMA, f32 accumulate, fully fused
// expansion. Weights (3 MB bf16) + input (12.8 MB) are L2-resident; HBM sees
// only the ~31 MB compulsory traffic.
// ---------------------------------------------------------------------------

typedef __attribute__((ext_vector_type(16))) __bf16       v16bf;
typedef __attribute__((ext_vector_type(8)))  float        v8f;
typedef __attribute__((ext_vector_type(4)))  unsigned int u32x4;

#define NGROUP   32
#define FEAT     164     // real features per group
#define KCHUNKS  6       // padded K per group = 6 * 32 = 192
#define BN       32      // spatial positions per block (3136 % 32 == 0)
#define HW       3136    // 56*56
#define FSTR     200     // LDS stride (bf16) for feature rows (16B aligned)

#if defined(__gfx1250__) && \
    __has_builtin(__builtin_amdgcn_global_load_async_to_lds_b32) && \
    __has_builtin(__builtin_amdgcn_s_wait_asynccnt)
#define USE_ASYNC_LDS 1
typedef __attribute__((address_space(1))) int as1_int;
typedef __attribute__((address_space(3))) int as3_int;
#else
#define USE_ASYNC_LDS 0
#endif

static __device__ __forceinline__ unsigned short f32_to_bf16_rne(float x) {
    union { float f; unsigned int u; } v; v.f = x;
    unsigned int u = v.u;
    u += ((u >> 16) & 1u) + 0x7FFFu;           // round-to-nearest-even
    return (unsigned short)(u >> 16);
}

// ---------------------------------------------------------------------------
// Pre-pass: fp32 weight -> bf16, WMMA-tiled layout wb[g][kc][o][kk]
//   element index = ((g*6 + kc)*256 + o)*32 + kk ; f = kc*32+kk, zero-padded
// ---------------------------------------------------------------------------
__global__ __launch_bounds__(256) void convert_weight_kernel(
        const float* __restrict__ w, unsigned short* __restrict__ wb) {
    int idx  = blockIdx.x * 256 + threadIdx.x;         // 32*6*256*32 = 1572864
    int kk   = idx & 31;
    int o    = (idx >> 5) & 255;
    int rest = idx >> 13;                              // g*6 + kc
    int kc   = rest % 6;
    int g    = rest / 6;
    int f    = kc * 32 + kk;
    float val = (f < FEAT) ? w[(size_t)o * (NGROUP * FEAT) + g * FEAT + f] : 0.0f;
    wb[idx] = f32_to_bf16_rne(val);
}

// ---------------------------------------------------------------------------
// Fused expansion + GEMM. One block = 8 waves = 256 threads.
// Block tile: M=256 x N=32. Wave w owns M rows [32w,32w+32): 2x2 wmma tiles.
// Pipeline per group g (one s_barrier each):
//   expand(g+1) from LDS input slab  -> feature buffer (g+1)&1
//   async-stage input slab for g+2   -> gin buffer g&1 (hidden under WMMA)
//   WMMA chain over feature buffer g&1
//   s_wait_asynccnt 0 ; s_barrier
// ---------------------------------------------------------------------------
__global__ __launch_bounds__(256) void poly_gemm_kernel(
        const float* __restrict__ in,            // (4,256,56,56)
        const unsigned short* __restrict__ wb,   // bf16 tiled weights (3 MB)
        const float* __restrict__ bias,          // (256,)
        const int* __restrict__ pm2,             // (36,2)
        const int* __restrict__ pcm,             // (120,2)
        float* __restrict__ out) {               // (4,256,56,56)

    __shared__ unsigned short feat[2 * BN * FSTR];    // 2 x 12800 B
    __shared__ float gin[2][8 * BN];                  // 2 x 1 KB input slabs
    __shared__ int spm2a[36], spm2b[36], spcma[120], spcmb[120];

    const int tid  = threadIdx.x;
    const int lane = tid & 31;
    const int wav  = tid >> 5;          // 0..7
    const int ncol = lane & 15;         // row (A) / col (B,D) within a 16-tile
    const int khal = lane >> 4;         // K-half selector per ISA 16-bit layout

    const int p0  = blockIdx.x * BN;
    const int b   = p0 / HW;
    const int hw0 = p0 % HW;

    // Per-thread element of the 8x32 input slab: channel c = tid>>5, pos n = tid&31.
    // LDS destination is simply gin[buf] + tid (contiguous).
    const float* gsrc_base =
        in + ((size_t)b * 256 + (size_t)(tid >> 5)) * HW + hw0 + (tid & 31);

    auto stage_gin = [&](int g, int buf) {
        const float* src = gsrc_base + (size_t)g * 8 * HW;
#if USE_ASYNC_LDS
        int* srcg = (int*)const_cast<float*>(src);     // generic, non-const
        int* dstl = (int*)&gin[buf][tid];              // generic (LDS object)
        __builtin_amdgcn_global_load_async_to_lds_b32(
            (as1_int*)srcg, (as3_int*)dstl, 0, 0);
#else
        gin[buf][tid] = *src;
#endif
    };

    // stage index tables
    if (tid < 36)       { spm2a[tid] = pm2[tid * 2];  spm2b[tid] = pm2[tid * 2 + 1]; }
    else if (tid < 156) { int j = tid - 36;
                          spcma[j] = pcm[j * 2];      spcmb[j] = pcm[j * 2 + 1]; }

    // ---- polynomial expansion of one group into an LDS bf16 tile ----------
    // Each thread: position n = lane, features f = wav + 8*i, i = 0..23.
    auto expand = [&](int buf, unsigned short* fb) {
        float v[8];
        #pragma unroll
        for (int k = 0; k < 8; ++k) v[k] = gin[buf][k * BN + lane];  // conflict-free

        unsigned short* fr = fb + lane * FSTR;
        #pragma unroll
        for (int i = 0; i < 24; ++i) {
            int f = wav + i * 8;
            float val;
            if (f < 8) {
                val = v[f];
            } else if (f < 44) {
                int j = f - 8;
                val = v[spm2a[j]] * v[spm2b[j]];
            } else if (f < FEAT) {
                int j = f - 44;
                int q = spcmb[j];
                val = v[spcma[j]] * (v[spm2a[q]] * v[spm2b[q]]);
            } else {
                val = 0.0f;                          // K padding
            }
            fr[f] = f32_to_bf16_rne(val);
        }
    };

    v8f acc[2][2];
    #pragma unroll
    for (int mt = 0; mt < 2; ++mt)
        #pragma unroll
        for (int nt = 0; nt < 2; ++nt)
            acc[mt][nt] = (v8f)0.0f;

    // ---- prologue: slab(0) -> expand feat(0); slab(1) staged --------------
    stage_gin(0, 0);
#if USE_ASYNC_LDS
    __builtin_amdgcn_s_wait_asynccnt(0);
#endif
    __syncthreads();                       // tables + gin[0] visible
    expand(0, feat);
    stage_gin(1, 1);
#if USE_ASYNC_LDS
    __builtin_amdgcn_s_wait_asynccnt(0);
#endif
    __syncthreads();                       // feat(0) + gin[1] visible

    for (int g = 0; g < NGROUP; ++g) {
        // expand NEXT group (reads gin[(g+1)&1], staged in the previous iteration);
        // its VALU + ds_stores co-execute with the WMMA chain below.
        if (g + 1 < NGROUP) {
            __builtin_prefetch(wb + (size_t)(g + 1) * KCHUNKS * 256 * 32, 0, 0);
            expand((g + 1) & 1, feat + ((g + 1) & 1) * BN * FSTR);
        }
        // async-stage slab for g+2 into gin[g&1] (its last reader finished
        // before the barrier that ended iteration g-1).
        if (g + 2 < NGROUP) stage_gin(g + 2, g & 1);

        const unsigned short* fbase = feat + (g & 1) * BN * FSTR;

        #pragma unroll
        for (int kc = 0; kc < KCHUNKS; ++kc) {
            // A (weights): row m = 32*wav + 16*mt + ncol ; ISA K-pair layout
            const unsigned short* arow =
                wb + ((size_t)(g * KCHUNKS + kc) * 256 + wav * 32 + ncol) * 32;
            union { u32x4 q[2]; v16bf v; } a0, a1, b0, b1;
            a0.q[0] = *(const u32x4*)(arow + 8 * khal);
            a0.q[1] = *(const u32x4*)(arow + 16 + 8 * khal);
            a1.q[0] = *(const u32x4*)(arow + 16 * 32 + 8 * khal);
            a1.q[1] = *(const u32x4*)(arow + 16 * 32 + 16 + 8 * khal);

            // B (features): col n = 16*nt + ncol ; contiguous K in LDS row
            const unsigned short* brow0 = fbase + ncol * FSTR + kc * 32 + 8 * khal;
            const unsigned short* brow1 = fbase + (16 + ncol) * FSTR + kc * 32 + 8 * khal;
            b0.q[0] = *(const u32x4*)(brow0);
            b0.q[1] = *(const u32x4*)(brow0 + 16);
            b1.q[0] = *(const u32x4*)(brow1);
            b1.q[1] = *(const u32x4*)(brow1 + 16);

            acc[0][0] = __builtin_amdgcn_wmma_f32_16x16x32_bf16(
                false, a0.v, false, b0.v, (short)0, acc[0][0], false, false);
            acc[0][1] = __builtin_amdgcn_wmma_f32_16x16x32_bf16(
                false, a0.v, false, b1.v, (short)0, acc[0][1], false, false);
            acc[1][0] = __builtin_amdgcn_wmma_f32_16x16x32_bf16(
                false, a1.v, false, b0.v, (short)0, acc[1][0], false, false);
            acc[1][1] = __builtin_amdgcn_wmma_f32_16x16x32_bf16(
                false, a1.v, false, b1.v, (short)0, acc[1][1], false, false);
        }

#if USE_ASYNC_LDS
        __builtin_amdgcn_s_wait_asynccnt(0);   // drain this wave's async store
#endif
        __syncthreads();                        // one barrier per group
    }

    // ---- epilogue: add bias, store fp32 -----------------------------------
    float* outb = out + (size_t)b * 256 * HW + hw0;
    #pragma unroll
    for (int mt = 0; mt < 2; ++mt) {
        #pragma unroll
        for (int r = 0; r < 8; ++r) {
            int m = wav * 32 + mt * 16 + 8 * khal + r;    // ISA C/D row layout
            float bv = bias[m];
            #pragma unroll
            for (int nt = 0; nt < 2; ++nt) {
                int n = nt * 16 + ncol;
                outb[(size_t)m * HW + n] = acc[mt][nt][r] + bv;
            }
        }
    }
}

// ---------------------------------------------------------------------------
// Host launcher
// ---------------------------------------------------------------------------
extern "C" void kernel_launch(void* const* d_in, const int* in_sizes, int n_in,
                              void* d_out, int out_size, void* d_ws, size_t ws_size,
                              hipStream_t stream) {
    const float* input  = (const float*)d_in[0];
    const float* weight = (const float*)d_in[1];
    const float* bias   = (const float*)d_in[2];
    const int*   pm2    = (const int*)d_in[3];
    const int*   pcm    = (const int*)d_in[4];
    float*       outp   = (float*)d_out;

    unsigned short* wb = (unsigned short*)d_ws;   // 32*6*256*32*2 = 3 MB

    const int wb_elems = NGROUP * KCHUNKS * 256 * 32;    // 1572864
    convert_weight_kernel<<<wb_elems / 256, 256, 0, stream>>>(weight, wb);

    const int nblocks = (4 * HW) / BN;                   // 392
    poly_gemm_kernel<<<nblocks, 256, 0, stream>>>(input, wb, bias, pm2, pcm, outp);
}
